// PointViewFusion_26757646254389
// MI455X (gfx1250) — matrix operand order repaired and checked by reading, not compiled, not run on hardware
//
#include <hip/hip_runtime.h>

typedef __bf16 bf16;
typedef __attribute__((ext_vector_type(16))) __bf16 v16bf;
typedef __attribute__((ext_vector_type(8)))  float  v8f;

#define HD    256
#define FUSD  512
#define SMAX  512
#define BD    8
#define NPTS  16384
#define NTOT  (BD * NPTS)      // 131072 points
#define NSEG  (BD * SMAX + 1)  // 4097 segments (incl. dump)

constexpr int KTILE = 32;          // bf16 WMMA K per instruction
constexpr int WAVES = 8;           // 256 threads
constexpr int BLK_M = WAVES * 16;  // 128 rows per block

// ---------------------------------------------------------------------------
// One-shot weight pack: W [K][NOUT] f32 -> Wt [NOUT][K] bf16 (transposed so a
// K-chunk of one output column is 64 contiguous bytes -> async b128 staging).
// ---------------------------------------------------------------------------
__global__ void __launch_bounds__(256)
pack_wt(const float* __restrict__ W, bf16* __restrict__ Wt, int K, int NOUT) {
  long i = (long)blockIdx.x * 256 + threadIdx.x;
  if (i < (long)K * NOUT) {
    int n = (int)(i / K);
    int k = (int)(i - (long)n * K);
    Wt[i] = (bf16)W[(long)k * NOUT + n];
  }
}

// Issue async global->LDS copies for W chunk [*, k0:k0+32]: NOUT rows x 64B.
template <int NOUT>
__device__ inline void stage_async(const bf16* __restrict__ Wt, int K, int k0,
                                   bf16* dst, int tid) {
  constexpr int WORDS = NOUT * 4;  // 128-bit words per chunk
#pragma unroll
  for (int w = tid; w < WORDS; w += 256) {
    const int n = w >> 2, part = w & 3;
    uint32_t lds = (uint32_t)(uintptr_t)(dst + n * KTILE + part * 8);
    uint64_t ga  = (uint64_t)(uintptr_t)(Wt + (size_t)n * K + k0 + part * 8);
    asm volatile("global_load_async_to_lds_b128 %0, %1, off"
                 :: "v"(lds), "v"(ga) : "memory");
  }
}

__device__ inline void wait_async0() {
  asm volatile("s_wait_asynccnt 0x0" ::: "memory");
}

// ---------------------------------------------------------------------------
// Fused GEMM + LayerNorm epilogue, bf16 WMMA (v_wmma_f32_16x16x32_bf16),
// double-buffered async-to-LDS weight staging.
// EPI: 0 = LN+ReLU, store bf16   (optional visibility MASK)
//      1 = LN+ReLU, fold into attn = sigmoid(h @ w2 + b2), store f32 [N]
//      2 = bias only, store f32
//      3 = LN (no relu), store f32  (ASCALE: scale A cols >= K/2 by refined[row])
// ---------------------------------------------------------------------------
template <int K, int NOUT, bool ABF16, int EPI, bool MASK, bool ASCALE>
__global__ void __launch_bounds__(256)
gemm_ln(const void* __restrict__ Av, const bf16* __restrict__ Wt,
        const float* __restrict__ bias, const float* __restrict__ gamma,
        const float* __restrict__ beta, const int* __restrict__ mask,
        const float* __restrict__ w2, const float* __restrict__ b2,
        const float* __restrict__ refined,
        bf16* __restrict__ outB, float* __restrict__ outF,
        int ostride, int ooffs) {
  static_assert(K % KTILE == 0 && NOUT % 16 == 0, "tiling");
  constexpr int NT = NOUT / 16;
  __shared__ __align__(16) bf16 ldsW[2][KTILE * NOUT];  // [n][kk], double buffer

  const int tid  = threadIdx.x;
  const int wave = tid >> 5;
  const int lane = tid & 31;
  const int h    = lane >> 4;     // half-group selects K subset
  const int ln16 = lane & 15;
  const long row0  = (long)blockIdx.x * BLK_M;
  const long myrow = row0 + wave * 16 + ln16;   // A-fragment row for this lane

  const float* Af = (const float*)Av;
  const bf16*  Ab = (const bf16*)Av;

  float scale2 = 1.0f;
  if (ASCALE) scale2 = refined[myrow];

  v8f acc[NT];
#pragma unroll
  for (int t = 0; t < NT; ++t) acc[t] = (v8f){0, 0, 0, 0, 0, 0, 0, 0};

  // prologue: stage first chunk
  stage_async<NOUT>(Wt, K, 0, &ldsW[0][0], tid);
  wait_async0();
  __syncthreads();

  int cur = 0;
  for (int k0 = 0; k0 < K; k0 += KTILE) {
    // stream next chunk into the other buffer while we compute
    if (k0 + KTILE < K)
      stage_async<NOUT>(Wt, K, k0 + KTILE, &ldsW[cur ^ 1][0], tid);

    // A fragment: 16x32 bf16 (ISA layout: lane-half h holds K = (e<8?e:e+8)+8h)
    v16bf a;
    {
      const long base = myrow * (long)K;
      const int  c1 = k0 + 8 * h;
      const int  c2 = k0 + 16 + 8 * h;
#pragma unroll
      for (int e = 0; e < 8; ++e) {
        int   c = c1 + e;
        float v = ABF16 ? (float)Ab[base + c] : Af[base + c];
        if (ASCALE && c >= K / 2) v *= scale2;
        a[e] = (bf16)v;
      }
#pragma unroll
      for (int e = 0; e < 8; ++e) {
        int   c = c2 + e;
        float v = ABF16 ? (float)Ab[base + c] : Af[base + c];
        if (ASCALE && c >= K / 2) v *= scale2;
        a[8 + e] = (bf16)v;
      }
    }

#pragma unroll
    for (int t = 0; t < NT; ++t) {
      // B fragment: 32x16 bf16; lane group g holds K = (e<8?e:e+8)+8g, col=ln16
      v16bf b;
      const bf16* src = &ldsW[cur][(t * 16 + ln16) * KTILE + 8 * h];
#pragma unroll
      for (int e = 0; e < 16; ++e) b[e] = src[(e < 8) ? e : e + 8];
      acc[t] = __builtin_amdgcn_wmma_f32_16x16x32_bf16(
          false, a, false, b, (short)0, acc[t], false, false);
    }

    wait_async0();     // our async writes into buf^1 are done
    __syncthreads();   // everyone finished reading buf; writes visible
    cur ^= 1;
  }

  // ---------------- epilogue: bias + LayerNorm stats in registers ------------
  float sum[8], sq[8];
#pragma unroll
  for (int r = 0; r < 8; ++r) { sum[r] = 0.0f; sq[r] = 0.0f; }
#pragma unroll
  for (int t = 0; t < NT; ++t) {
    float bv = bias[t * 16 + ln16];
#pragma unroll
    for (int r = 0; r < 8; ++r) {
      float x = acc[t][r] + bv;
      acc[t][r] = x;
      sum[r] += x;
      sq[r]  += x * x;
    }
  }

  if (EPI == 2) {  // bias only -> f32
#pragma unroll
    for (int t = 0; t < NT; ++t)
#pragma unroll
      for (int r = 0; r < 8; ++r) {
        long row = row0 + wave * 16 + r + 8 * h;   // C/D: M = r + 8*halfgroup
        outF[row * (long)ostride + ooffs + t * 16 + ln16] = acc[t][r];
      }
    return;
  }

  // row reduction: butterflies stay inside the 16-lane half-group
#pragma unroll
  for (int r = 0; r < 8; ++r) {
#pragma unroll
    for (int m = 1; m < 16; m <<= 1) {
      sum[r] += __shfl_xor(sum[r], m, 32);
      sq[r]  += __shfl_xor(sq[r],  m, 32);
    }
  }
  const float invN = 1.0f / (float)NOUT;
  float mean[8], rstd[8];
#pragma unroll
  for (int r = 0; r < 8; ++r) {
    mean[r]   = sum[r] * invN;
    float var = sq[r] * invN - mean[r] * mean[r];
    rstd[r]   = rsqrtf(var + 1e-5f);
  }

  float attp[8];
#pragma unroll
  for (int r = 0; r < 8; ++r) attp[r] = 0.0f;

#pragma unroll
  for (int t = 0; t < NT; ++t) {
    const int   col = t * 16 + ln16;
    const float g   = gamma[col];
    const float be  = beta[col];
    const float w2v = (EPI == 1) ? w2[col] : 0.0f;
#pragma unroll
    for (int r = 0; r < 8; ++r) {
      float y = (acc[t][r] - mean[r]) * rstd[r] * g + be;
      if (EPI == 0 || EPI == 1) y = fmaxf(y, 0.0f);
      long row = row0 + wave * 16 + r + 8 * h;
      if (MASK) y *= (float)mask[row];
      if (EPI == 1) {
        attp[r] += y * w2v;
      } else if (EPI == 0) {
        outB[row * (long)ostride + ooffs + col] = (bf16)y;
      } else {  // EPI == 3
        outF[row * (long)ostride + ooffs + col] = y;
      }
    }
  }

  if (EPI == 1) {
#pragma unroll
    for (int r = 0; r < 8; ++r)
#pragma unroll
      for (int m = 1; m < 16; m <<= 1) attp[r] += __shfl_xor(attp[r], m, 32);
    if (ln16 == 0) {
      const float bb = b2[0];
#pragma unroll
      for (int r = 0; r < 8; ++r) {
        long row = row0 + wave * 16 + r + 8 * h;
        outF[row] = 1.0f / (1.0f + expf(-(attp[r] + bb)));
      }
    }
  }
}

// ---------------------------------------------------------------------------
__global__ void __launch_bounds__(256) zero_f32(float* p, long n) {
  long i = (long)blockIdx.x * 256 + threadIdx.x;
  if (i < n) p[i] = 0.0f;
}

// one wave per point: accumulate segment count / attn / geo sums
__global__ void __launch_bounds__(256)
scatter_seg(const float* __restrict__ geo, const float* __restrict__ attn,
            const int* __restrict__ spid, float* __restrict__ cnt,
            float* __restrict__ suma, float* __restrict__ sumg) {
  const int p    = blockIdx.x * 8 + (threadIdx.x >> 5);
  const int lane = threadIdx.x & 31;
  const int id   = spid[p];
  const int bidx = p / NPTS;
  const int seg  = (id >= 0) ? (bidx * SMAX + id) : (BD * SMAX);
  if (lane == 0) {
    atomicAdd(&cnt[seg], 1.0f);
    atomicAdd(&suma[seg], attn[p]);
  }
  const float* g  = geo  + (long)p   * HD;
  float*       sg = sumg + (long)seg * HD;
  for (int c = lane; c < HD; c += 32) atomicAdd(&sg[c], g[c]);
}

// one wave per point: cosine sim with segment mean -> refined attention
__global__ void __launch_bounds__(256)
refine_pts(const float* __restrict__ geo, const float* __restrict__ attn,
           const int* __restrict__ spid, const float* __restrict__ cnt,
           const float* __restrict__ suma, const float* __restrict__ sumg,
           float* __restrict__ refined) {
  const int  p     = blockIdx.x * 8 + (threadIdx.x >> 5);
  const int  lane  = threadIdx.x & 31;
  const int  id    = spid[p];
  const int  bidx  = p / NPTS;
  const bool valid = (id >= 0);
  const int  seg   = valid ? (bidx * SMAX + id) : (BD * SMAX);
  const float inv  = 1.0f / fmaxf(cnt[seg], 1.0f);
  const float* g   = geo  + (long)p   * HD;
  const float* sg  = sumg + (long)seg * HD;
  float num = 0.0f, g2 = 0.0f, m2 = 0.0f;
  for (int c = lane; c < HD; c += 32) {
    float gv = g[c];
    float mv = sg[c] * inv;
    num += gv * mv;
    g2  += gv * gv;
    m2  += mv * mv;
  }
#pragma unroll
  for (int m = 1; m < 32; m <<= 1) {
    num += __shfl_xor(num, m, 32);
    g2  += __shfl_xor(g2,  m, 32);
    m2  += __shfl_xor(m2,  m, 32);
  }
  if (lane == 0) {
    float a   = attn[p];
    float ma  = suma[seg] * inv;
    float den = fmaxf(sqrtf(g2) * sqrtf(m2), 1e-8f);
    float sim = num / den;
    refined[p] = valid ? (ma + 0.2f * (a - ma) * sim) : a;
  }
}

// ---------------------------------------------------------------------------
extern "C" void kernel_launch(void* const* d_in, const int* in_sizes, int n_in,
                              void* d_out, int out_size, void* d_ws,
                              size_t ws_size, hipStream_t stream) {
  const float* pf    = (const float*)d_in[0];
  const float* vf    = (const float*)d_in[1];
  const int*   spid  = (const int*)d_in[2];
  const int*   vmask = (const int*)d_in[3];
  const float* pp_w = (const float*)d_in[4],  *pp_b = (const float*)d_in[5];
  const float* pp_g = (const float*)d_in[6],  *pp_be = (const float*)d_in[7];
  const float* vp_w = (const float*)d_in[8],  *vp_b = (const float*)d_in[9];
  const float* vp_g = (const float*)d_in[10], *vp_be = (const float*)d_in[11];
  const float* at_w1 = (const float*)d_in[12], *at_b1 = (const float*)d_in[13];
  const float* at_g  = (const float*)d_in[14], *at_be = (const float*)d_in[15];
  const float* at_w2 = (const float*)d_in[16], *at_b2 = (const float*)d_in[17];
  const float* ge_w1 = (const float*)d_in[18], *ge_b1 = (const float*)d_in[19];
  const float* ge_g  = (const float*)d_in[20], *ge_be = (const float*)d_in[21];
  const float* ge_w2 = (const float*)d_in[22], *ge_b2 = (const float*)d_in[23];
  const float* fu_w  = (const float*)d_in[24], *fu_b  = (const float*)d_in[25];
  const float* fu_g  = (const float*)d_in[26], *fu_be = (const float*)d_in[27];

  // workspace carve-up
  char* ws = (char*)d_ws;
  bf16* cat = (bf16*)ws;   ws += (size_t)NTOT * (2 * HD) * sizeof(bf16);  // [N,512]
  bf16* gh  = (bf16*)ws;   ws += (size_t)NTOT * HD * sizeof(bf16);        // geo hidden
  float* geo = (float*)ws; ws += (size_t)NTOT * HD * sizeof(float);       // geo [N,256]
  float* attn = (float*)ws;    ws += (size_t)NTOT * sizeof(float);
  float* refined = (float*)ws; ws += (size_t)NTOT * sizeof(float);
  float* cnt  = (float*)ws;    ws += (size_t)NSEG * sizeof(float);
  float* suma = (float*)ws;    ws += (size_t)NSEG * sizeof(float);
  float* sumg = (float*)ws;    ws += (size_t)NSEG * HD * sizeof(float);
  // packed bf16 transposed weights [NOUT][K]
  bf16* WtPP = (bf16*)ws;  ws += (size_t)256 * 256 * sizeof(bf16);
  bf16* WtVP = (bf16*)ws;  ws += (size_t)256 * 768 * sizeof(bf16);
  bf16* WtAT = (bf16*)ws;  ws += (size_t)256 * 512 * sizeof(bf16);
  bf16* WtG1 = (bf16*)ws;  ws += (size_t)256 * 512 * sizeof(bf16);
  bf16* WtG2 = (bf16*)ws;  ws += (size_t)256 * 256 * sizeof(bf16);
  bf16* WtFU = (bf16*)ws;  ws += (size_t)512 * 512 * sizeof(bf16);

  const dim3 blk(256);
  const dim3 grd(NTOT / BLK_M);   // 1024 blocks

  // pack weights -> bf16 [NOUT][K]
  pack_wt<<<(256 * 256 + 255) / 256, blk, 0, stream>>>(pp_w, WtPP, 256, 256);
  pack_wt<<<(768 * 256 + 255) / 256, blk, 0, stream>>>(vp_w, WtVP, 768, 256);
  pack_wt<<<(512 * 256 + 255) / 256, blk, 0, stream>>>(at_w1, WtAT, 512, 256);
  pack_wt<<<(512 * 256 + 255) / 256, blk, 0, stream>>>(ge_w1, WtG1, 512, 256);
  pack_wt<<<(256 * 256 + 255) / 256, blk, 0, stream>>>(ge_w2, WtG2, 256, 256);
  pack_wt<<<(512 * 512 + 255) / 256, blk, 0, stream>>>(fu_w, WtFU, 512, 512);

  // pp = relu(LN(point @ pp_w + b)) -> cat[:, 0:256] (bf16)
  gemm_ln<256, 256, false, 0, false, false><<<grd, blk, 0, stream>>>(
      pf, WtPP, pp_b, pp_g, pp_be, nullptr, nullptr, nullptr, nullptr,
      cat, nullptr, 2 * HD, 0);
  // vp = relu(LN(view @ vp_w + b)) * mask -> cat[:, 256:512]
  gemm_ln<768, 256, false, 0, true, false><<<grd, blk, 0, stream>>>(
      vf, WtVP, vp_b, vp_g, vp_be, vmask, nullptr, nullptr, nullptr,
      cat, nullptr, 2 * HD, HD);
  // attn = sigmoid(relu(LN(cat @ att_w1 + b)) @ att_w2 + b2)
  gemm_ln<512, 256, true, 1, false, false><<<grd, blk, 0, stream>>>(
      cat, WtAT, at_b1, at_g, at_be, nullptr, at_w2, at_b2, nullptr,
      nullptr, attn, 0, 0);
  // gh = relu(LN(cat @ geo_w1 + b))  (bf16)
  gemm_ln<512, 256, true, 0, false, false><<<grd, blk, 0, stream>>>(
      cat, WtG1, ge_b1, ge_g, ge_be, nullptr, nullptr, nullptr, nullptr,
      gh, nullptr, HD, 0);
  // geo = gh @ geo_w2 + b2  (f32)
  gemm_ln<256, 256, true, 2, false, false><<<grd, blk, 0, stream>>>(
      gh, WtG2, ge_b2, nullptr, nullptr, nullptr, nullptr, nullptr, nullptr,
      nullptr, geo, HD, 0);

  // segment reductions (zero accumulators each call for determinism)
  const long nz = (long)NSEG * (HD + 2);  // cnt + suma + sumg are contiguous
  zero_f32<<<(int)((nz + 255) / 256), blk, 0, stream>>>(cnt, nz);
  scatter_seg<<<NTOT / 8, blk, 0, stream>>>(geo, attn, spid, cnt, suma, sumg);
  refine_pts<<<NTOT / 8, blk, 0, stream>>>(geo, attn, spid, cnt, suma, sumg,
                                           refined);

  // Z = LN([pp, vp*refined] @ fus_w + b)  -> d_out (f32)
  gemm_ln<512, 512, true, 3, false, true><<<grd, blk, 0, stream>>>(
      cat, WtFU, fu_b, fu_g, fu_be, nullptr, nullptr, nullptr, refined,
      nullptr, (float*)d_out, FUSD, 0);
}
